// MultiHeadAttention_77137612636263
// MI455X (gfx1250) — matrix operand order repaired and checked by reading, not compiled, not run on hardware
//
#include <hip/hip_runtime.h>
#include <cstdint>
#include <cstddef>

// ---------------------------------------------------------------------------
// Sliding-window MHA for MI455X (gfx1250, wave32).
// bf16 WMMA (v_wmma_f32_16x16x32_bf16) for all three GEMM stages, fp32
// accumulation, flash-style online softmax for the banded attention.
// GEMMs stage K-slabs in LDS via async global->LDS copies (ASYNCcnt path).
// ---------------------------------------------------------------------------

typedef float  v8f   __attribute__((ext_vector_type(8)));
typedef __bf16 v16bf __attribute__((ext_vector_type(16)));
typedef __bf16 bf8v  __attribute__((ext_vector_type(8)));
typedef int    v4i   __attribute__((ext_vector_type(4)));

#define B_    2
#define L_    2048
#define D_    768
#define E_    768
#define H_    12
#define HD_   64
#define N3E_  2304
#define HWIN_ 128          // WIN/2

__device__ __forceinline__ v16bf cat16(bf8v lo, bf8v hi) {
  return __builtin_shufflevector(lo, hi, 0,1,2,3,4,5,6,7,8,9,10,11,12,13,14,15);
}

__device__ __forceinline__ v8f wmma_bf16(v16bf a, v16bf b, v8f c) {
  // D = A(16x32 bf16) x B(32x16 bf16) + C(16x16 f32)
  return __builtin_amdgcn_wmma_f32_16x16x32_bf16(false, a, false, b,
                                                 (short)0, c, false, false);
}

// reductions within each 16-lane half of a wave32 (xor masks 1..8 stay in group)
__device__ __forceinline__ float rmax16(float v) {
#pragma unroll
  for (int m = 1; m < 16; m <<= 1) v = fmaxf(v, __shfl_xor(v, m, 32));
  return v;
}
__device__ __forceinline__ float rsum16(float v) {
#pragma unroll
  for (int m = 1; m < 16; m <<= 1) v += __shfl_xor(v, m, 32);
  return v;
}

// ---- async global -> LDS copy of one 16-byte chunk ------------------------
#if defined(__has_builtin)
#if __has_builtin(__builtin_amdgcn_global_load_async_to_lds_b128)
#define HAS_ASYNC_LDS 1
#endif
#endif

typedef __attribute__((address_space(1))) v4i* gptr4;   // global int4*
typedef __attribute__((address_space(3))) v4i* lptr4;   // LDS int4*

__device__ __forceinline__ void async_copy16(const __bf16* g, __bf16* l) {
#ifdef HAS_ASYNC_LDS
  __builtin_amdgcn_global_load_async_to_lds_b128(
      (gptr4)(v4i*)const_cast<__bf16*>(g), (lptr4)(v4i*)l, 0, 0);
#else
  *(bf8v*)l = *(const bf8v*)g;   // fallback: relay through VGPRs
#endif
}

__device__ __forceinline__ void async_wait_and_barrier() {
#ifdef HAS_ASYNC_LDS
#if __has_builtin(__builtin_amdgcn_s_wait_asynccnt)
  __builtin_amdgcn_s_wait_asynccnt(0);
#else
  asm volatile("s_wait_asynccnt 0x0" ::: "memory");
#endif
#endif
  __syncthreads();
}

// ---------------------------------------------------------------------------
// fp32 -> bf16 conversion
// ---------------------------------------------------------------------------
__global__ void cvt_bf16_kernel(const float* __restrict__ src,
                                __bf16* __restrict__ dst, int n) {
  int i = blockIdx.x * blockDim.x + threadIdx.x;
  if (i < n) dst[i] = (__bf16)src[i];
}

// ---------------------------------------------------------------------------
// GEMM: C[M,N] = A[M,K] (bf16, row major) x W[N,K]^T (bf16, row major) + bias
// mode 0: scatter into q [B,H,L,64], k [B,H,L,64], vT [B,H,64,L]  (bf16)
// mode 1: write fp32 out[M,N]
// Block: 128 threads = 4 waves arranged 2(M) x 2(N); wave tile 64x32
// (8 WMMA accumulators). Block tile 128(M) x 64(N); K-slabs of A (128x32)
// and W (64x32) staged in LDS via async global->LDS copies.
// ---------------------------------------------------------------------------
__global__ __launch_bounds__(128) void gemm_bf16_kernel(
    const __bf16* __restrict__ A, const __bf16* __restrict__ W,
    const float* __restrict__ bias, int M, int N, int K, int mode,
    __bf16* __restrict__ qb, __bf16* __restrict__ kb, __bf16* __restrict__ vtb,
    float* __restrict__ out) {
  __shared__ __align__(16) __bf16 asl[128 * 32];   // 8 KB
  __shared__ __align__(16) __bf16 wsl[64 * 32];    // 4 KB

  const int tid  = threadIdx.x;
  const int wave = tid >> 5;
  const int lane = tid & 31;
  const int lh   = lane >> 4;     // lane half (0/1)
  const int ln   = lane & 15;
  const int waveM = (wave >> 1) * 64;   // 0 / 64
  const int waveN = (wave & 1) * 32;    // 0 / 32
  const int rowBase = blockIdx.x * 128;
  const int colBase = blockIdx.y * 64;

  v8f acc[4][2] = {};

  for (int k0 = 0; k0 < K; k0 += 32) {
    // ---- stage K-slabs: A 128x32 (512 x 16B), W 64x32 (256 x 16B) -------
#pragma unroll
    for (int iss = 0; iss < 4; ++iss) {
      int c   = iss * 128 + tid;        // 0..511
      int row = c >> 2;
      int kc  = (c & 3) * 8;
      async_copy16(A + (size_t)(rowBase + row) * K + k0 + kc,
                   asl + row * 32 + kc);
    }
#pragma unroll
    for (int iss = 0; iss < 2; ++iss) {
      int c  = iss * 128 + tid;         // 0..255
      int n  = c >> 2;
      int kc = (c & 3) * 8;
      async_copy16(W + (size_t)(colBase + n) * K + k0 + kc,
                   wsl + n * 32 + kc);
    }
    async_wait_and_barrier();

    // ---- fragments from LDS + 8 WMMAs ----------------------------------
    v16bf bf[2];
#pragma unroll
    for (int j = 0; j < 2; ++j) {
      const __bf16* p = wsl + (waveN + j * 16 + ln) * 32 + lh * 8;
      bf[j] = cat16(*(const bf8v*)p, *(const bf8v*)(p + 16));
    }
#pragma unroll
    for (int t = 0; t < 4; ++t) {
      const __bf16* p = asl + (waveM + t * 16 + ln) * 32 + lh * 8;
      v16bf af = cat16(*(const bf8v*)p, *(const bf8v*)(p + 16));
#pragma unroll
      for (int j = 0; j < 2; ++j)
        acc[t][j] = wmma_bf16(af, bf[j], acc[t][j]);
    }
    __syncthreads();   // all reads done before next slab overwrites LDS
  }

  // ---- epilogue --------------------------------------------------------
#pragma unroll
  for (int t = 0; t < 4; ++t)
#pragma unroll
    for (int j = 0; j < 2; ++j)
#pragma unroll
      for (int r = 0; r < 8; ++r) {
        int row = rowBase + waveM + t * 16 + r + 8 * lh;  // C layout M
        int col = colBase + waveN + j * 16 + ln;          // C layout N
        float v = acc[t][j][r] + bias[col];
        if (mode == 0) {
          int h = col / 192, rr = col % 192;
          int chunk = rr / 64, d = rr % 64;
          int bb = row / L_, rl = row % L_;
          size_t bh = (size_t)bb * H_ + h;
          if (chunk == 0)       qb [(bh * L_ + rl) * HD_ + d] = (__bf16)v;
          else if (chunk == 1)  kb [(bh * L_ + rl) * HD_ + d] = (__bf16)v;
          else                  vtb[(bh * HD_ + d) * L_ + rl] = (__bf16)v;
        } else {
          out[(size_t)row * N + col] = v;
        }
      }
}

// ---------------------------------------------------------------------------
// Banded flash attention. 1 wave = one 16-row Q tile of one (b,h).
// Block = 128 threads = 4 independent waves; per-wave LDS stages P (16x32 bf16)
// to convert C-layout probabilities into A-fragment layout for the PV WMMAs.
// ---------------------------------------------------------------------------
__global__ __launch_bounds__(128) void attn_kernel(
    const __bf16* __restrict__ qb, const __bf16* __restrict__ kb,
    const __bf16* __restrict__ vtb, const int* __restrict__ pad,
    __bf16* __restrict__ vals) {
  __shared__ __align__(16) __bf16 lds[4][16 * 32];

  const int wave = threadIdx.x >> 5;
  const int lane = threadIdx.x & 31;
  const int lh   = lane >> 4;
  const int ln   = lane & 15;
  const int tile = blockIdx.x * 4 + wave;   // 0 .. B*H*(L/16)-1
  const int bh   = tile >> 7;               // / (L/16)
  const int m0   = (tile & 127) * 16;
  const int b    = bh / H_;
  const int h    = bh % H_;

  const __bf16* qbase = qb  + (size_t)bh * L_ * HD_;
  const __bf16* kbase = kb  + (size_t)bh * L_ * HD_;
  const __bf16* vbase = vtb + (size_t)bh * HD_ * L_;
  __bf16* plds = &lds[wave][0];

  // Q A-fragments for both 32-wide head-dim chunks
  v16bf aq[2];
#pragma unroll
  for (int c = 0; c < 2; ++c) {
    const __bf16* p = qbase + (size_t)(m0 + ln) * HD_ + c * 32 + lh * 8;
    aq[c] = cat16(*(const bf8v*)p, *(const bf8v*)(p + 16));
  }

  float mrow[8], srow[8];
  int rowpad[8];
#pragma unroll
  for (int r = 0; r < 8; ++r) {
    mrow[r] = -__builtin_inff();
    srow[r] = 0.f;
    rowpad[r] = pad[b * L_ + m0 + r + 8 * lh];
  }
  v8f o[4] = {};

  int jstart = m0 - HWIN_; if (jstart < 0) jstart = 0;
  jstart &= ~31;
  int jend = m0 + 16 + HWIN_; if (jend > L_) jend = L_;   // L_ % 32 == 0

  for (int j0 = jstart; j0 < jend; j0 += 32) {
    // ---- scores for two 16-key subtiles --------------------------------
    v8f s[2];
    int cpad[2];
#pragma unroll
    for (int sub = 0; sub < 2; ++sub) {
      int col = j0 + sub * 16 + ln;          // < L_ by construction
      cpad[sub] = pad[b * L_ + col];
      v8f sc = {};
#pragma unroll
      for (int c = 0; c < 2; ++c) {
        const __bf16* p = kbase + (size_t)col * HD_ + c * 32 + lh * 8;
        v16bf bk = cat16(*(const bf8v*)p, *(const bf8v*)(p + 16));
        sc = wmma_bf16(aq[c], bk, sc);
      }
      s[sub] = sc;
    }

    // ---- online softmax update ----------------------------------------
    float fac[8];
#pragma unroll
    for (int r = 0; r < 8; ++r) {
      int rg = m0 + r + 8 * lh;
      int c0 = j0 + ln, c1 = j0 + 16 + ln;
      float v0 = (float)s[0][r] * 0.125f;    // 1/sqrt(64)
      float v1 = (float)s[1][r] * 0.125f;
      int d0 = rg - c0; if (d0 < 0) d0 = -d0;
      int d1 = rg - c1; if (d1 < 0) d1 = -d1;
      if (d0 > HWIN_ || cpad[0] || rowpad[r]) v0 = -__builtin_inff();
      if (d1 > HWIN_ || cpad[1] || rowpad[r]) v1 = -__builtin_inff();

      float tm   = rmax16(fmaxf(v0, v1));
      float mnew = fmaxf(mrow[r], tm);
      float f, e0, e1;
      if (mnew == -__builtin_inff()) { f = 0.f; e0 = 0.f; e1 = 0.f; }
      else {
        f  = __expf(mrow[r] - mnew);   // expf(-inf)=0 handles first tile
        e0 = __expf(v0 - mnew);
        e1 = __expf(v1 - mnew);
      }
      srow[r] = srow[r] * f + rsum16(e0 + e1);
      mrow[r] = mnew;
      fac[r]  = f;
      // stash probs (C layout -> LDS row major 16x32)
      plds[(r + 8 * lh) * 32 + ln]      = (__bf16)e0;
      plds[(r + 8 * lh) * 32 + 16 + ln] = (__bf16)e1;
    }
#pragma unroll
    for (int a = 0; a < 4; ++a)
#pragma unroll
      for (int r = 0; r < 8; ++r) o[a][r] *= fac[r];

    // reload P as A fragment (row = ln, K split 8+8 per lane half)
    const __bf16* pp = plds + ln * 32 + lh * 8;
    v16bf ap = cat16(*(const bf8v*)pp, *(const bf8v*)(pp + 16));

    // ---- PV: O += P(16x32) x V(32x64) ---------------------------------
#pragma unroll
    for (int a = 0; a < 4; ++a) {
      const __bf16* vp = vbase + (size_t)(a * 16 + ln) * L_ + j0 + lh * 8;
      v16bf bv = cat16(*(const bf8v*)vp, *(const bf8v*)(vp + 16));
      o[a] = wmma_bf16(ap, bv, o[a]);
    }
  }

  // ---- normalize + store bf16 vals [B, L, E] ---------------------------
  float inv[8];
#pragma unroll
  for (int r = 0; r < 8; ++r) inv[r] = srow[r] > 0.f ? 1.f / srow[r] : 0.f;
#pragma unroll
  for (int a = 0; a < 4; ++a)
#pragma unroll
    for (int r = 0; r < 8; ++r) {
      int rl = m0 + r + 8 * lh;
      float v = o[a][r] * inv[r];
      vals[((size_t)b * L_ + rl) * E_ + h * HD_ + a * 16 + ln] = (__bf16)v;
    }
}

// ---------------------------------------------------------------------------
// launch
// ---------------------------------------------------------------------------
extern "C" void kernel_launch(void* const* d_in, const int* in_sizes, int n_in,
                              void* d_out, int out_size, void* d_ws, size_t ws_size,
                              hipStream_t stream) {
  const float* x    = (const float*)d_in[0];
  const int*   pad  = (const int*)  d_in[1];
  const float* Wqkv = (const float*)d_in[2];
  const float* bqkv = (const float*)d_in[3];
  const float* Wo   = (const float*)d_in[4];
  const float* bo   = (const float*)d_in[5];
  float* out = (float*)d_out;

  const size_t XB   = (size_t)B_ * L_ * D_;     // 3,145,728
  const size_t WQKV = (size_t)N3E_ * D_;        // 1,769,472
  const size_t WOX  = (size_t)E_ * E_;          //   589,824
  const size_t QKV1 = (size_t)B_ * H_ * L_ * HD_;
  const size_t VALS = (size_t)B_ * L_ * E_;

  char* ws = (char*)d_ws;
  size_t off = 0;
  __bf16* xb    = (__bf16*)(ws + off); off += XB   * 2;
  __bf16* wqkvb = (__bf16*)(ws + off); off += WQKV * 2;
  __bf16* wob   = (__bf16*)(ws + off); off += WOX  * 2;
  __bf16* qb    = (__bf16*)(ws + off); off += QKV1 * 2;
  __bf16* kb    = (__bf16*)(ws + off); off += QKV1 * 2;
  __bf16* vtb   = (__bf16*)(ws + off); off += QKV1 * 2;
  __bf16* valsb = (__bf16*)(ws + off); off += VALS * 2;   // ~34.5 MB total

  cvt_bf16_kernel<<<(int)((XB   + 255) / 256), 256, 0, stream>>>(x,    xb,    (int)XB);
  cvt_bf16_kernel<<<(int)((WQKV + 255) / 256), 256, 0, stream>>>(Wqkv, wqkvb, (int)WQKV);
  cvt_bf16_kernel<<<(int)((WOX  + 255) / 256), 256, 0, stream>>>(Wo,   wob,   (int)WOX);

  // QKV projection: [4096 x 768] x [768 x 2304]
  gemm_bf16_kernel<<<dim3(4096 / 128, N3E_ / 64), 128, 0, stream>>>(
      xb, wqkvb, bqkv, B_ * L_, N3E_, D_, 0, qb, kb, vtb, nullptr);

  // banded attention: B*H*(L/16) = 3072 wave tiles, 4 per block
  attn_kernel<<<(B_ * H_ * (L_ / 16)) / 4, 128, 0, stream>>>(qb, kb, vtb, pad, valsb);

  // output projection: [4096 x 768] x [768 x 768] -> fp32 out
  gemm_bf16_kernel<<<dim3(4096 / 128, E_ / 64), 128, 0, stream>>>(
      valsb, wob, bo, B_ * L_, E_, E_, 1, nullptr, nullptr, nullptr, out);
}
